// GCN_63445256896634
// MI455X (gfx1250) — compile-verified, matrix operand order checked by reading, not run on hardware
//
#include <hip/hip_runtime.h>
#include <hip/hip_bf16.h>

typedef __attribute__((ext_vector_type(16))) _Float16 v16h;
typedef __attribute__((ext_vector_type(8)))  float    v8f;

// ---------------------------------------------------------------------------
// WMMA 16x16x32 f16 lane layout helper (CDNA5 ISA §7.12.2):
// 16-bit A 16x32: lanes 0-15 hold M=lane, VGPR0..3 = K 0..7, VGPR4..7 = K16..23
//                 lanes 16-31 hold M=lane-16, K 8..15 and K 24..31.
__device__ __forceinline__ int klocal(int lane, int j) {
    return ((lane >> 4) << 3) + (j < 8 ? j : j + 8);
}

// ---------------------------------------------------------------------------
// Stage 0: degree / dinv / per-edge norm
// ---------------------------------------------------------------------------
__global__ __launch_bounds__(256) void k_deg_init(float* __restrict__ deg, int n) {
    int i = blockIdx.x * blockDim.x + threadIdx.x;
    if (i < n) deg[i] = 1.0f;                        // self loop
}

__global__ __launch_bounds__(256) void k_deg_edges(const int* __restrict__ ei,
                                                   float* __restrict__ deg, int E) {
    int e = blockIdx.x * blockDim.x + threadIdx.x;
    if (e < E) atomicAdd(&deg[ei[E + e]], 1.0f);     // dst = ei[1][e]
}

__global__ __launch_bounds__(256) void k_dinv(const float* __restrict__ deg,
                                              float* __restrict__ dinv, int n) {
    int i = blockIdx.x * blockDim.x + threadIdx.x;
    if (i < n) dinv[i] = rsqrtf(deg[i]);             // deg >= 1 always
}

__global__ __launch_bounds__(256) void k_norm(const int* __restrict__ ei,
                                              const float* __restrict__ dinv,
                                              float* __restrict__ norm, int E) {
    int e = blockIdx.x * blockDim.x + threadIdx.x;
    if (e < E) norm[e] = dinv[ei[e]] * dinv[ei[E + e]];
}

// ---------------------------------------------------------------------------
// Stage 1: pre-swizzle W1 / W2 into WMMA B-fragment lane layout (f16)
// ---------------------------------------------------------------------------
__global__ __launch_bounds__(256) void k_prep_w(const float* __restrict__ W1,
                                                const float* __restrict__ W2,
                                                _Float16* __restrict__ w1B,
                                                _Float16* __restrict__ w2B) {
    int t = blockIdx.x * blockDim.x + threadIdx.x;
    if (t < 4 * 32 * 16) {
        int c = t / (32 * 16);
        int l = (t / 16) & 31;
        int j = t & 15;
        int k = 32 * c + klocal(l, j);
        w1B[t] = (_Float16)W1[k * 16 + (l & 15)];
    } else if (t < 4 * 32 * 16 + 32 * 16) {
        int u = t - 4 * 32 * 16;
        int l = u / 16;
        int j = u & 15;
        int k = klocal(l, j);
        int n = l & 15;
        w2B[u] = (k < 16 && n < 7) ? (_Float16)W2[k * 7 + n] : (_Float16)0.0f;
    }
}

// ---------------------------------------------------------------------------
// Stage 2: h = x @ W1 (100000x128 @ 128x16), one wave per 16-row tile,
// 4x v_wmma_f32_16x16x32_f16.  Epilogue fuses the self-loop seed:
// acc1[i][c] = h[i][c] * dinv[i]^2.
// ---------------------------------------------------------------------------
__global__ __launch_bounds__(256) void k_gemm1(const float* __restrict__ x,
                                               const _Float16* __restrict__ w1B,
                                               const float* __restrict__ dinv,
                                               float* __restrict__ h,
                                               float* __restrict__ acc1, int nrows) {
    int wave = (blockIdx.x * blockDim.x + threadIdx.x) >> 5;  // wave32
    int lane = threadIdx.x & 31;
    int row0 = wave * 16;
    if (row0 >= nrows) return;                                // uniform per wave
    int rsub = lane & 15;
    int hi   = lane >> 4;                                     // 0 or 1
    const float* xr = x + (size_t)(row0 + rsub) * 128 + hi * 8;
    v8f c = {};
#pragma unroll
    for (int kc = 0; kc < 4; ++kc) {
        const float* p = xr + kc * 32;
        v16h a;
#pragma unroll
        for (int j = 0; j < 8; ++j) a[j] = (_Float16)p[j];          // K base+0..7
#pragma unroll
        for (int j = 0; j < 8; ++j) a[8 + j] = (_Float16)p[16 + j]; // K base+16..23
        v16h b = *(const v16h*)(w1B + (size_t)(kc * 32 + lane) * 16);
        c = __builtin_amdgcn_wmma_f32_16x16x32_f16(false, a, false, b,
                                                   (short)0, c, false, false);
    }
    // D layout: VGPR r -> M = row0 + hi*8 + r, N = rsub
#pragma unroll
    for (int r = 0; r < 8; ++r) {
        int row = row0 + hi * 8 + r;
        float hv = c[r];
        float dv = dinv[row];                  // broadcast across 16 lanes
        h   [(size_t)row * 16 + rsub] = hv;
        acc1[(size_t)row * 16 + rsub] = hv * dv * dv;
    }
}

// ---------------------------------------------------------------------------
// Stage 3: layer-1 edge scatter (4 threads/edge, float4 gather + 4 atomics)
// ---------------------------------------------------------------------------
__global__ __launch_bounds__(256) void k_scatter1(const int* __restrict__ ei,
                                                  const float* __restrict__ norm,
                                                  const float* __restrict__ h,
                                                  float* __restrict__ acc1, int E) {
    int t = blockIdx.x * blockDim.x + threadIdx.x;
    if (t >= 4 * E) return;
    int e = t >> 2, q = t & 3;
    int s = ei[e], d = ei[E + e];
    float nm = norm[e];
    const float4 v = *(const float4*)(h + (size_t)s * 16 + q * 4);
    float* ap = acc1 + (size_t)d * 16 + q * 4;
    atomicAdd(ap + 0, v.x * nm);
    atomicAdd(ap + 1, v.y * nm);
    atomicAdd(ap + 2, v.z * nm);
    atomicAdd(ap + 3, v.w * nm);
}

// ---------------------------------------------------------------------------
// Stage 4: h2 = relu(acc1 + b1) @ W2  (bias+relu fused into A-fragment load)
// K = 16 padded to 32, N = 7 padded to 16. One WMMA per 16-row tile.
// Epilogue fuses the layer-2 self-loop seed: acc2 = h2 * dinv^2.
// h2/acc2 stride 8; padded column 7 is exactly 0.
// ---------------------------------------------------------------------------
__global__ __launch_bounds__(256) void k_gemm2(const float* __restrict__ acc1,
                                               const float* __restrict__ b1,
                                               const _Float16* __restrict__ w2B,
                                               const float* __restrict__ dinv,
                                               float* __restrict__ h2,
                                               float* __restrict__ acc2, int nrows) {
    int wave = (blockIdx.x * blockDim.x + threadIdx.x) >> 5;
    int lane = threadIdx.x & 31;
    int row0 = wave * 16;
    if (row0 >= nrows) return;
    int rsub = lane & 15;
    int hi   = lane >> 4;
    int base = hi * 8;                               // lane's K offset (0 or 8)
    const float* p = acc1 + (size_t)(row0 + rsub) * 16 + base;
    v16h a;
#pragma unroll
    for (int j = 0; j < 8; ++j) {
        float f = p[j] + b1[base + j];
        a[j] = (_Float16)fmaxf(f, 0.0f);             // relu(layer1)
    }
#pragma unroll
    for (int j = 8; j < 16; ++j) a[j] = (_Float16)0.0f;  // K padding 16..31
    v16h b = *(const v16h*)(w2B + (size_t)lane * 16);
    v8f c = {};
    c = __builtin_amdgcn_wmma_f32_16x16x32_f16(false, a, false, b,
                                               (short)0, c, false, false);
    if (rsub < 8) {                                  // cols 0..6 + zero pad col 7
#pragma unroll
        for (int r = 0; r < 8; ++r) {
            int row = row0 + base + r;
            float hv = c[r];
            float dv = dinv[row];
            h2  [(size_t)row * 8 + rsub] = hv;
            acc2[(size_t)row * 8 + rsub] = hv * dv * dv;
        }
    }
}

// ---------------------------------------------------------------------------
// Stage 5: layer-2 edge scatter + final bias/relu
// ---------------------------------------------------------------------------
__global__ __launch_bounds__(256) void k_scatter2(const int* __restrict__ ei,
                                                  const float* __restrict__ norm,
                                                  const float* __restrict__ h2,
                                                  float* __restrict__ acc2, int E) {
    int t = blockIdx.x * blockDim.x + threadIdx.x;
    if (t >= 2 * E) return;
    int e = t >> 1, q = t & 1;
    int s = ei[e], d = ei[E + e];
    float nm = norm[e];
    const float4 v = *(const float4*)(h2 + (size_t)s * 8 + q * 4);
    float* ap = acc2 + (size_t)d * 8 + q * 4;
    atomicAdd(ap + 0, v.x * nm);
    atomicAdd(ap + 1, v.y * nm);
    atomicAdd(ap + 2, v.z * nm);
    atomicAdd(ap + 3, v.w * nm);                     // q==1 lane 3 -> zero pad col
}

__global__ __launch_bounds__(256) void k_final(const float* __restrict__ acc2,
                                               const float* __restrict__ b2,
                                               float* __restrict__ out, int n7) {
    int t = blockIdx.x * blockDim.x + threadIdx.x;
    if (t < n7) {
        int i = t / 7, c = t % 7;
        out[t] = fmaxf(acc2[(size_t)i * 8 + c] + b2[c], 0.0f);
    }
}

// ---------------------------------------------------------------------------
extern "C" void kernel_launch(void* const* d_in, const int* in_sizes, int n_in,
                              void* d_out, int out_size, void* d_ws, size_t ws_size,
                              hipStream_t stream) {
    const float* x  = (const float*)d_in[0];
    const int*   ei = (const int*)d_in[1];       // [2, E] (harness int convention)
    const float* W1 = (const float*)d_in[2];
    const float* b1 = (const float*)d_in[3];
    const float* W2 = (const float*)d_in[4];
    const float* b2 = (const float*)d_in[5];
    float* out = (float*)d_out;

    const int N = in_sizes[0] / 128;             // 100000
    const int E = in_sizes[1] / 2;               // 3200000

    // workspace layout (all offsets 32B aligned)
    char* ws = (char*)d_ws;
    float*    h    = (float*)(ws + 0);                       // N*16 f32  (6.4 MB)
    float*    acc1 = (float*)(ws + (size_t)N * 16 * 4);      // N*16 f32
    float*    h2   = (float*)(ws + (size_t)N * 32 * 4);      // N*8  f32
    float*    acc2 = (float*)(ws + (size_t)N * 40 * 4);      // N*8  f32
    float*    deg  = (float*)(ws + (size_t)N * 48 * 4);      // N f32
    float*    dinv = (float*)(ws + (size_t)N * 49 * 4);      // N f32
    float*    norm = (float*)(ws + (size_t)N * 50 * 4);      // E f32 (12.8 MB)
    _Float16* w1B  = (_Float16*)(ws + (size_t)N * 50 * 4 + (size_t)E * 4);        // 4 KB
    _Float16* w2B  = (_Float16*)(ws + (size_t)N * 50 * 4 + (size_t)E * 4 + 4096); // 1 KB

    const int B = 256;
    auto blk = [](long long t, int b) { return (int)((t + b - 1) / b); };

    // degree / norm
    k_deg_init <<<blk(N, B), B, 0, stream>>>(deg, N);
    k_deg_edges<<<blk(E, B), B, 0, stream>>>(ei, deg, E);
    k_dinv     <<<blk(N, B), B, 0, stream>>>(deg, dinv, N);
    k_norm     <<<blk(E, B), B, 0, stream>>>(ei, dinv, norm, E);

    // weight swizzle for WMMA B fragments
    k_prep_w<<<blk(4 * 32 * 16 + 32 * 16, B), B, 0, stream>>>(W1, W2, w1B, w2B);

    // layer 1
    int waves1 = (N + 15) / 16;
    k_gemm1   <<<blk((long long)waves1 * 32, B), B, 0, stream>>>(x, w1B, dinv, h, acc1, N);
    k_scatter1<<<blk((long long)E * 4, B), B, 0, stream>>>(ei, norm, h, acc1, E);

    // layer 2
    k_gemm2   <<<blk((long long)waves1 * 32, B), B, 0, stream>>>(acc1, b1, w2B, dinv, h2, acc2, N);
    k_scatter2<<<blk((long long)E * 2, B), B, 0, stream>>>(ei, norm, h2, acc2, E);

    // bias + relu -> out
    k_final<<<blk((long long)N * 7, B), B, 0, stream>>>(acc2, b2, out, N * 7);
}
